// MyQSlink_Model_85907935855377
// MI455X (gfx1250) — compile-verified, matrix operand order checked by reading, not run on hardware
//
#include <hip/hip_runtime.h>
#include <hip/hip_bf16.h>

typedef __attribute__((ext_vector_type(16))) _Float16 v16h;
typedef __attribute__((ext_vector_type(8)))  float    v8f;
typedef __attribute__((ext_vector_type(4)))  unsigned int u32x4;
typedef __attribute__((ext_vector_type(4)))  int i32x4;
typedef __attribute__((ext_vector_type(8)))  int i32x8;

#define HID   768
#define NLAY  12
#define NHEAD 12
#define DH    64
#define FFDIM 3072
#define BSZ   16
#define SL    256
#define NTOK  (BSZ*SL)   // 4096

#define BKT 32
#define LPAD 8
#define LSTR (BKT + LPAD)

// ---------------- TDM support (guarded; falls back to reg-pipeline) -------
#if defined(__has_builtin)
#  if __has_builtin(__builtin_amdgcn_tensor_load_to_lds)
#    define HAVE_TDM 1
#  endif
#endif
#ifndef HAVE_TDM
#  define HAVE_TDM 0
#endif

#if HAVE_TDM
#  if __has_builtin(__builtin_amdgcn_s_wait_tensorcnt)
#    define TENSOR_WAIT(n) __builtin_amdgcn_s_wait_tensorcnt((short)(n))
#  else
#    define TENSOR_WAIT(n) asm volatile("s_wait_tensorcnt %0" :: "i"(n) : "memory")
#  endif
#  if __clang_major__ >= 23
#    define TDM_LOAD(g0, g1, z4, z8) __builtin_amdgcn_tensor_load_to_lds((g0), (g1), (z4), (z4), (z8), 0)
#  else
#    define TDM_LOAD(g0, g1, z4, z8) __builtin_amdgcn_tensor_load_to_lds((g0), (g1), (z4), (z4), 0)
#  endif

// D# group1: data_size=2B, pad_enable, pad_interval=16 dwords (code 3),
// pad_amount=4 dwords (code 3)  -> LDS row stride = 32 halves + 8 halves pad = LSTR
__device__ __forceinline__ i32x8 tdm_desc_g1(int len0, int len1, int rows, long long stride) {
  i32x8 d;
  d[0] = (int)((1u << 16) | (1u << 20) | (3u << 22) | (3u << 25));
  d[1] = (int)(((unsigned)len0 & 0xFFFFu) << 16);                              // dim0[15:0]
  d[2] = (int)(((unsigned)len0 >> 16) | (((unsigned)len1 & 0xFFFFu) << 16));   // dim0[31:16] | dim1[15:0]
  d[3] = (int)(((unsigned)len1 >> 16) | ((unsigned)BKT << 16));                // dim1[31:16] | tile_dim0
  d[4] = rows;                                                                 // tile_dim1 | tile_dim2=0
  d[5] = (int)(unsigned)(stride & 0xFFFFFFFFll);                               // dim0_stride[31:0]
  d[6] = (int)((stride >> 32) & 0xFFFFll);                                     // dim0_stride[47:32]
  d[7] = 0;
  return d;
}

__device__ __forceinline__ void tdm_tile_load(const _Float16* gsrc, unsigned lds_off, i32x8 g1) {
  const unsigned long long ga = (unsigned long long)(size_t)gsrc;
  u32x4 g0;
  g0.x = 1u;                                                   // count=1 (user D#)
  g0.y = lds_off;                                              // LDS byte address
  g0.z = (unsigned)ga;                                         // global_addr[31:0]
  g0.w = (unsigned)((ga >> 32) & 0x1FFFFFFull) | (2u << 30);   // global_addr[56:32] | type=2
  i32x4 z4 = {0, 0, 0, 0};
  i32x8 z8 = {0, 0, 0, 0, 0, 0, 0, 0};
  (void)z8;
  TDM_LOAD(g0, g1, z4, z8);
}
#endif  // HAVE_TDM

// ---------------- WMMA tile compute (shared by both staging paths) -------
template <int WM, int NWN, int MF>
__device__ __forceinline__ void wmma_tile_compute(
    const _Float16 (*__restrict__ Asb)[LSTR], const _Float16 (*__restrict__ Bsb)[LSTR],
    v8f (&acc)[MF][4], int lane, int wm, int wn) {
  const int koff = (lane >> 4) * 8;
  union F16V { v16h v; u32x4 u[2]; };
  F16V af[MF];
#pragma unroll
  for (int mf = 0; mf < MF; mf++) {
    const int ar = wm * WM + mf * 16 + (lane & 15);
    af[mf].u[0] = *(const u32x4*)&Asb[ar][koff];
    af[mf].u[1] = *(const u32x4*)&Asb[ar][16 + koff];
  }
#pragma unroll
  for (int nt = 0; nt < 4; nt++) {
    const int bc = wn * 64 + nt * 16 + (lane & 15);
    F16V bf;
    bf.u[0] = *(const u32x4*)&Bsb[bc][koff];
    bf.u[1] = *(const u32x4*)&Bsb[bc][16 + koff];
#pragma unroll
    for (int mf = 0; mf < MF; mf++)
      acc[mf][nt] = __builtin_amdgcn_wmma_f32_16x16x32_f16(
          false, af[mf].v, false, bf.v, (short)0, acc[mf][nt], false, false);
  }
}

// ---------------- WMMA GEMM (B always N-major: Bt[N][K]) ----------------
// C[M,N] = alpha * A[M,K] x Bt[N,K]^T + bias[N]
// fp16 in, fp32 accumulate. TDM double-buffered LDS staging when available;
// otherwise software-pipelined global->reg->LDS. Batched via blockIdx.z.
template <int WM, int NWN>
__global__ __launch_bounds__(256) void wmma_gemm_nk(
    const _Float16* __restrict__ A, const _Float16* __restrict__ Bt,
    float* __restrict__ C, const float* __restrict__ bias,
    int M, int N, int K, int lda, int ldb, int ldc, float alpha,
    int zInner, long long aO, long long aI, long long bO, long long bI,
    long long cO, long long cI)
{
  constexpr int BMT = 128;
  constexpr int BNT = NWN * 64;
  constexpr int MF  = WM / 16;

  const int z = blockIdx.z;
  const _Float16* Ab = A  + (long long)(z / zInner) * aO + (long long)(z % zInner) * aI;
  const _Float16* Bb = Bt + (long long)(z / zInner) * bO + (long long)(z % zInner) * bI;
  float*          Cb = C  + (long long)(z / zInner) * cO + (long long)(z % zInner) * cI;

  const int m0 = blockIdx.y * BMT;
  const int n0 = blockIdx.x * BNT;
  const int tid  = threadIdx.x;
  const int lane = tid & 31;
  const int wave = tid >> 5;
  const int wm = wave / NWN;
  const int wn = wave % NWN;

  v8f acc[MF][4] = {};

#if HAVE_TDM
  __shared__ __align__(16) _Float16 As[2][BMT][LSTR];
  __shared__ __align__(16) _Float16 Bs[2][BNT][LSTR];
  const unsigned ldsA[2] = {(unsigned)(size_t)&As[0][0][0], (unsigned)(size_t)&As[1][0][0]};
  const unsigned ldsB[2] = {(unsigned)(size_t)&Bs[0][0][0], (unsigned)(size_t)&Bs[1][0][0]};
  const _Float16* aTile = Ab + (long long)m0 * lda;
  const _Float16* bTile = Bb + (long long)n0 * ldb;
  const i32x8 gA = tdm_desc_g1(K, M, BMT, (long long)lda);
  const i32x8 gB = tdm_desc_g1(K, N, BNT, (long long)ldb);

  if (wave == 0) {
    tdm_tile_load(aTile, ldsA[0], gA);
    tdm_tile_load(bTile, ldsB[0], gB);
  }
  int buf = 0;
  for (int k0 = 0; k0 < K; k0 += BKT, buf ^= 1) {
    if (wave == 0) {
      if (k0 + BKT < K) {   // kick DMA for next tile, then wait for current
        tdm_tile_load(aTile + k0 + BKT, ldsA[buf ^ 1], gA);
        tdm_tile_load(bTile + k0 + BKT, ldsB[buf ^ 1], gB);
        TENSOR_WAIT(2);
      } else {
        TENSOR_WAIT(0);
      }
    }
    __syncthreads();
    wmma_tile_compute<WM, NWN, MF>(As[buf], Bs[buf], acc, lane, wm, wn);
    __syncthreads();
  }
#else
  __shared__ __align__(16) _Float16 As[BMT][LSTR];
  __shared__ __align__(16) _Float16 Bs[BNT][LSTR];
  constexpr int BREGS = (BNT * BKT) / (256 * 8);

  const int a_row = tid >> 1;
  const int a_seg = (tid & 1) * 16;
  const int b_row = (BNT == 128) ? (tid >> 1) : (tid >> 2);
  const int b_seg = (BNT == 128) ? ((tid & 1) * 16) : ((tid & 3) * 8);

  const _Float16* aPtr = Ab + (long long)(m0 + a_row) * lda + a_seg;
  const _Float16* bPtr = Bb + (long long)(n0 + b_row) * ldb + b_seg;

  u32x4 aR[2], bR[2];
  aR[0] = ((const u32x4*)aPtr)[0];
  aR[1] = ((const u32x4*)aPtr)[1];
  bR[0] = ((const u32x4*)bPtr)[0];
  if constexpr (BREGS == 2) bR[1] = ((const u32x4*)bPtr)[1];

  for (int k0 = 0; k0 < K; k0 += BKT) {
    *(u32x4*)&As[a_row][a_seg]     = aR[0];
    *(u32x4*)&As[a_row][a_seg + 8] = aR[1];
    *(u32x4*)&Bs[b_row][b_seg]     = bR[0];
    if constexpr (BREGS == 2) *(u32x4*)&Bs[b_row][b_seg + 8] = bR[1];
    __syncthreads();

    if (k0 + BKT < K) {
      const _Float16* pa = aPtr + k0 + BKT;
      const _Float16* pb = bPtr + k0 + BKT;
      aR[0] = ((const u32x4*)pa)[0];
      aR[1] = ((const u32x4*)pa)[1];
      bR[0] = ((const u32x4*)pb)[0];
      if constexpr (BREGS == 2) bR[1] = ((const u32x4*)pb)[1];
      if (k0 + 2 * BKT < K) {
        __builtin_prefetch(aPtr + k0 + 2 * BKT, 0, 1);
        __builtin_prefetch(bPtr + k0 + 2 * BKT, 0, 1);
      }
    }
    wmma_tile_compute<WM, NWN, MF>(As, Bs, acc, lane, wm, wn);
    __syncthreads();
  }
#endif

  // epilogue: C frag layout — lane<16: (M=r, N=lane); lane>=16: (M=r+8, N=lane-16)
#pragma unroll
  for (int mf = 0; mf < MF; mf++) {
    const int rbase = m0 + wm * WM + mf * 16 + (lane >> 4) * 8;
    const int cbase = n0 + wn * 64 + (lane & 15);
#pragma unroll
    for (int nt = 0; nt < 4; nt++) {
      const int col = cbase + nt * 16;
      const float bv = bias ? bias[col] : 0.f;
#pragma unroll
      for (int r = 0; r < 8; r++)
        Cb[(long long)(rbase + r) * ldc + col] = alpha * acc[mf][nt][r] + bv;
    }
  }
}

// ---------------- weight transpose + f32->f16 (LDS-tiled) ----------------
__global__ __launch_bounds__(256) void transpose_cvt_kernel(
    const float* __restrict__ W, _Float16* __restrict__ Wt, int K, int N) {
  __shared__ float tile[32][33];
  const int kb = blockIdx.x * 32, nb = blockIdx.y * 32;
  const int tx = threadIdx.x & 31, ty = threadIdx.x >> 5;
#pragma unroll
  for (int i = 0; i < 32; i += 8)
    tile[ty + i][tx] = W[(long long)(kb + ty + i) * N + nb + tx];
  __syncthreads();
#pragma unroll
  for (int i = 0; i < 32; i += 8)
    Wt[(long long)(nb + ty + i) * K + kb + tx] = (_Float16)tile[tx][ty + i];
}

__global__ void vtrans_kernel(const float* __restrict__ vf,
                              _Float16* __restrict__ vt) {
  const int bh = blockIdx.y, b = bh / NHEAD, h = bh % NHEAD;
  const int idx = blockIdx.x * 256 + threadIdx.x;
  const int n = idx >> 8, k = idx & 255;
  vt[((long long)bh * DH + n) * SL + k] =
      (_Float16)vf[((long long)b * SL + k) * HID + h * DH + n];
}

// ---------------- elementwise / small kernels ----------------
__global__ void f32_to_f16_kernel(const float* __restrict__ s,
                                  _Float16* __restrict__ d, long long n) {
  long long i = (long long)blockIdx.x * blockDim.x + threadIdx.x;
  if (i < n) d[i] = (_Float16)s[i];
}

__global__ void embed_kernel(const int* __restrict__ datas,
                             const float* __restrict__ we,
                             const float* __restrict__ pe,
                             const float* __restrict__ te,
                             float* __restrict__ x) {
  const int tok = blockIdx.x;
  const long long tokid = datas[tok];
  const int pos = tok % SL;
  for (int c = threadIdx.x; c < HID; c += blockDim.x)
    x[(long long)tok * HID + c] = we[tokid * HID + c] + pe[pos * HID + c] + te[c];
}

__global__ __launch_bounds__(256) void ln_kernel(
    const float* __restrict__ inp, const float* __restrict__ res,
    const float* __restrict__ g, const float* __restrict__ b,
    float* __restrict__ out) {
  const int row = blockIdx.x;
  const int tid = threadIdx.x;
  __shared__ float red[256];
  float v[3];
#pragma unroll
  for (int k = 0; k < 3; k++) {
    const int c = tid + k * 256;
    float t = inp[(long long)row * HID + c];
    if (res) t += res[(long long)row * HID + c];
    v[k] = t;
  }
  red[tid] = v[0] + v[1] + v[2];
  __syncthreads();
  for (int o = 128; o > 0; o >>= 1) { if (tid < o) red[tid] += red[tid + o]; __syncthreads(); }
  const float mean = red[0] / (float)HID;
  __syncthreads();
  float q = 0.f;
#pragma unroll
  for (int k = 0; k < 3; k++) { const float d = v[k] - mean; q += d * d; }
  red[tid] = q;
  __syncthreads();
  for (int o = 128; o > 0; o >>= 1) { if (tid < o) red[tid] += red[tid + o]; __syncthreads(); }
  const float rstd = rsqrtf(red[0] / (float)HID + 1e-12f);
#pragma unroll
  for (int k = 0; k < 3; k++) {
    const int c = tid + k * 256;
    out[(long long)row * HID + c] = (v[k] - mean) * rstd * g[c] + b[c];
  }
}

__global__ void gelu_kernel(float* __restrict__ x, long long n) {
  long long i = (long long)blockIdx.x * blockDim.x + threadIdx.x;
  if (i < n) {
    const float v = x[i];
    x[i] = 0.5f * v * (1.f + erff(v * 0.70710678118654752f));
  }
}

__global__ __launch_bounds__(256) void softmax_kernel(
    const float* __restrict__ S, _Float16* __restrict__ P) {
  const long long row = blockIdx.x;
  const int tid = threadIdx.x;
  __shared__ float red[256];
  const float v = S[row * SL + tid];
  red[tid] = v;
  __syncthreads();
  for (int o = 128; o > 0; o >>= 1) { if (tid < o) red[tid] = fmaxf(red[tid], red[tid + o]); __syncthreads(); }
  const float mx = red[0];
  __syncthreads();
  const float e = __expf(v - mx);
  red[tid] = e;
  __syncthreads();
  for (int o = 128; o > 0; o >>= 1) { if (tid < o) red[tid] += red[tid + o]; __syncthreads(); }
  P[row * SL + tid] = (_Float16)(e / red[0]);
}

__global__ void head_gemm_kernel(const float* __restrict__ X,
                                 const float* __restrict__ W,
                                 const float* __restrict__ bias,
                                 float* __restrict__ out, int T) {
  const int idx = blockIdx.x * blockDim.x + threadIdx.x;
  const int tok = idx / T, t = idx % T;
  if (tok >= NTOK) return;
  float s = bias[t];
  for (int k = 0; k < HID; k++) s += X[(long long)tok * HID + k] * W[k * T + t];
  out[(long long)tok * T + t] = s;
}

__global__ void zero_loss_kernel(float* o) { o[0] = 0.f; }

// ---------------- CRF ----------------
__global__ __launch_bounds__(512) void crf_llh_kernel(
    const float* __restrict__ em, const int* __restrict__ tags,
    const unsigned char* __restrict__ mask,
    const float* __restrict__ start, const float* __restrict__ endv,
    const float* __restrict__ trans, int T, float* __restrict__ loss) {
  __shared__ float alpha[BSZ][32];
  __shared__ float partial[BSZ];
  const int tid = threadIdx.x;
  const int b = tid >> 5, j = tid & 31;
  const bool act = (b < BSZ) && (j < T);
  if (act) alpha[b][j] = start[j] + em[((long long)b * SL) * T + j];
  __syncthreads();
  for (int t = 1; t < SL; t++) {
    float nxt = 0.f;
    if (act) {
      float mx = -1e30f;
      for (int i = 0; i < T; i++) mx = fmaxf(mx, alpha[b][i] + trans[i * T + j]);
      float ss = 0.f;
      for (int i = 0; i < T; i++) ss += __expf(alpha[b][i] + trans[i * T + j] - mx);
      nxt = mx + __logf(ss) + em[((long long)b * SL + t) * T + j];
      if (!mask[b * SL + t]) nxt = alpha[b][j];
    }
    __syncthreads();
    if (act) alpha[b][j] = nxt;
    __syncthreads();
  }
  if (act && j == 0) {
    float mx = -1e30f;
    for (int i = 0; i < T; i++) mx = fmaxf(mx, alpha[b][i] + endv[i]);
    float ss = 0.f;
    for (int i = 0; i < T; i++) ss += __expf(alpha[b][i] + endv[i] - mx);
    const float den = mx + __logf(ss);
    int prev = tags[b * SL];
    float score = start[prev] + em[((long long)b * SL) * T + prev];
    int lastidx = 0;
    for (int t = 1; t < SL; t++) {
      const int tg = tags[b * SL + t];
      const float m = mask[b * SL + t] ? 1.f : 0.f;
      score += m * (trans[prev * T + tg] + em[((long long)b * SL + t) * T + tg]);
      prev = tg;
      if (mask[b * SL + t]) lastidx = t;
    }
    partial[b] = score + endv[tags[b * SL + lastidx]] - den;
  }
  __syncthreads();
  if (tid == 0) {
    float s = 0.f;
    for (int bb = 0; bb < BSZ; bb++) s += partial[bb];
    atomicAdd(loss, -s);
  }
}

__global__ __launch_bounds__(512) void crf_decode_kernel(
    const float* __restrict__ em, const float* __restrict__ start,
    const float* __restrict__ endv, const float* __restrict__ trans, int T,
    unsigned char* __restrict__ hist, float* __restrict__ out) {
  __shared__ float alpha[BSZ][32];
  const int tid = threadIdx.x;
  const int b = tid >> 5, j = tid & 31;
  const bool act = (b < BSZ) && (j < T);
  if (act) alpha[b][j] = start[j] + em[((long long)b * SL) * T + j];
  __syncthreads();
  for (int t = 1; t < SL; t++) {
    float nxt = 0.f;
    if (act) {
      float mx = -1e30f; int am = 0;
      for (int i = 0; i < T; i++) {
        const float s = alpha[b][i] + trans[i * T + j];
        if (s > mx) { mx = s; am = i; }
      }
      nxt = mx + em[((long long)b * SL + t) * T + j];
      hist[((long long)(t - 1) * BSZ + b) * 32 + j] = (unsigned char)am;
    }
    __syncthreads();
    if (act) alpha[b][j] = nxt;
    __syncthreads();
  }
  if (act && j == 0) {
    float mx = -1e30f; int last = 0;
    for (int i = 0; i < T; i++) {
      const float s = alpha[b][i] + endv[i];
      if (s > mx) { mx = s; last = i; }
    }
    out[b * SL + (SL - 1)] = (float)last;
    int cur = last;
    for (int t = SL - 2; t >= 0; t--) {
      cur = hist[((long long)t * BSZ + b) * 32 + cur];
      out[b * SL + t] = (float)cur;
    }
  }
}

// ---------------- driver ----------------
extern "C" void kernel_launch(void* const* d_in, const int* in_sizes, int n_in,
                              void* d_out, int out_size, void* d_ws, size_t ws_size,
                              hipStream_t stream) {
  const int* datas            = (const int*)d_in[0];
  const int* ele_labels       = (const int*)d_in[1];
  const int* role_labels      = (const int*)d_in[2];
  const unsigned char* smask  = (const unsigned char*)d_in[3];
  const float* word_emb = (const float*)d_in[4];
  const float* pos_emb  = (const float*)d_in[5];
  const float* type_emb = (const float*)d_in[6];
  const float* emb_g = (const float*)d_in[7];
  const float* emb_b = (const float*)d_in[8];
  const float* Wq = (const float*)d_in[9];  const float* bq = (const float*)d_in[10];
  const float* Wk = (const float*)d_in[11]; const float* bk = (const float*)d_in[12];
  const float* Wv = (const float*)d_in[13]; const float* bv = (const float*)d_in[14];
  const float* Wo = (const float*)d_in[15]; const float* bo = (const float*)d_in[16];
  const float* ln1g = (const float*)d_in[17]; const float* ln1b = (const float*)d_in[18];
  const float* W1 = (const float*)d_in[19]; const float* b1 = (const float*)d_in[20];
  const float* W2 = (const float*)d_in[21]; const float* b2 = (const float*)d_in[22];
  const float* ln2g = (const float*)d_in[23]; const float* ln2b = (const float*)d_in[24];
  const float* W_ele  = (const float*)d_in[25]; const float* b_ele  = (const float*)d_in[26];
  const float* W_role = (const float*)d_in[27]; const float* b_role = (const float*)d_in[28];
  const float* ele_start = (const float*)d_in[29]; const float* ele_end = (const float*)d_in[30];
  const float* ele_trans = (const float*)d_in[31];
  const float* role_start = (const float*)d_in[32]; const float* role_end = (const float*)d_in[33];
  const float* role_trans = (const float*)d_in[34];
  (void)in_sizes; (void)n_in; (void)out_size; (void)ws_size;

  const long long U = (long long)NTOK * HID;             // 3.1M
  const long long S = (long long)BSZ * NHEAD * SL * SL;  // 12.58M (== NTOK*FFDIM)

  char* w = (char*)d_ws;
  size_t off = 0;
  auto alloc = [&](size_t bytes) -> void* {
    void* p = w + off;
    off = (off + bytes + 255) & ~(size_t)255;
    return p;
  };

  float* x    = (float*)alloc(U * 4);
  float* qf   = (float*)alloc(U * 4);
  float* kf   = (float*)alloc(U * 4);
  float* vf   = (float*)alloc(U * 4);
  float* cxf  = (float*)alloc(U * 4);
  float* aof  = (float*)alloc(U * 4);
  float* reg1 = (float*)alloc(S * 4);        // aliases: attn scores | FFN hidden (f32)
  float* sc = reg1;
  float* hf = reg1;
  float* ele_em  = (float*)alloc((long long)NTOK * 17 * 4);
  float* role_em = (float*)alloc((long long)NTOK * 7 * 4);
  _Float16* xh = (_Float16*)alloc(U * 2);
  _Float16* qh = (_Float16*)alloc(U * 2);
  _Float16* kh = (_Float16*)alloc(U * 2);
  _Float16* vt = (_Float16*)alloc(U * 2);    // per-head transposed V
  _Float16* ch = (_Float16*)alloc(U * 2);
  _Float16* reg2 = (_Float16*)alloc(S * 2);  // aliases: softmax probs | FFN hidden (f16)
  _Float16* ph = reg2;
  _Float16* hh = reg2;
  _Float16* wt = (_Float16*)alloc((long long)HID * FFDIM * 2);
  unsigned char* hist = (unsigned char*)alloc((long long)(SL - 1) * BSZ * 32);

  auto cvt = [&](const float* s, _Float16* d, long long n) {
    f32_to_f16_kernel<<<(int)((n + 255) / 256), 256, 0, stream>>>(s, d, n);
  };
  auto tcvt = [&](const float* Wsrc, _Float16* Wdst, int K, int N) {
    transpose_cvt_kernel<<<dim3(K / 32, N / 32), 256, 0, stream>>>(Wsrc, Wdst, K, N);
  };
  auto gemmBig = [&](const _Float16* A, const _Float16* Bm, float* C, const float* bias,
                     int M, int N, int K, int lda, int ldb, int ldc, float alpha,
                     int nz, int zInner, long long aO, long long aI, long long bO,
                     long long bI, long long cO, long long cI) {
    dim3 g(N / 128, M / 128, nz);
    wmma_gemm_nk<32, 2><<<g, 256, 0, stream>>>(A, Bm, C, bias, M, N, K, lda, ldb, ldc,
                                               alpha, zInner, aO, aI, bO, bI, cO, cI);
  };
  auto gemmSmall = [&](const _Float16* A, const _Float16* Bm, float* C, const float* bias,
                       int M, int N, int K, int lda, int ldb, int ldc, float alpha,
                       int nz, int zInner, long long aO, long long aI, long long bO,
                       long long bI, long long cO, long long cI) {
    dim3 g(N / 64, M / 128, nz);
    wmma_gemm_nk<16, 1><<<g, 256, 0, stream>>>(A, Bm, C, bias, M, N, K, lda, ldb, ldc,
                                               alpha, zInner, aO, aI, bO, bI, cO, cI);
  };

  // ---- embeddings + LN
  embed_kernel<<<NTOK, 256, 0, stream>>>(datas, word_emb, pos_emb, type_emb, x);
  ln_kernel<<<NTOK, 256, 0, stream>>>(x, nullptr, emb_g, emb_b, x);

  for (int i = 0; i < NLAY; i++) {
    const long long WO = (long long)i * HID * HID;
    cvt(x, xh, U);
    tcvt(Wq + WO, wt, HID, HID);
    gemmBig(xh, wt, qf, bq + i * HID, NTOK, HID, HID, HID, HID, HID, 1.f, 1, 1, 0, 0, 0, 0, 0, 0);
    tcvt(Wk + WO, wt, HID, HID);
    gemmBig(xh, wt, kf, bk + i * HID, NTOK, HID, HID, HID, HID, HID, 1.f, 1, 1, 0, 0, 0, 0, 0, 0);
    tcvt(Wv + WO, wt, HID, HID);
    gemmBig(xh, wt, vf, bv + i * HID, NTOK, HID, HID, HID, HID, HID, 1.f, 1, 1, 0, 0, 0, 0, 0, 0);
    cvt(qf, qh, U);
    cvt(kf, kh, U);
    vtrans_kernel<<<dim3(64, BSZ * NHEAD), 256, 0, stream>>>(vf, vt);
    // scores = (Q K^T)/sqrt(64), batched per (b,h); K tensor is already N-major
    gemmBig(qh, kh, sc, nullptr, SL, SL, DH, HID, HID, SL, 0.125f,
            BSZ * NHEAD, NHEAD,
            (long long)SL * HID, DH,
            (long long)SL * HID, DH,
            (long long)NHEAD * SL * SL, (long long)SL * SL);
    softmax_kernel<<<BSZ * NHEAD * SL, 256, 0, stream>>>(sc, ph);
    // ctx = P V   (B = per-head transposed V, N-major, ldb = SL)
    gemmSmall(ph, vt, cxf, nullptr, SL, DH, SL, SL, SL, HID, 1.f,
              BSZ * NHEAD, NHEAD,
              (long long)NHEAD * SL * SL, (long long)SL * SL,
              (long long)NHEAD * DH * SL, (long long)DH * SL,
              (long long)SL * HID, DH);
    cvt(cxf, ch, U);
    tcvt(Wo + WO, wt, HID, HID);
    gemmBig(ch, wt, aof, bo + i * HID, NTOK, HID, HID, HID, HID, HID, 1.f, 1, 1, 0, 0, 0, 0, 0, 0);
    ln_kernel<<<NTOK, 256, 0, stream>>>(aof, x, ln1g + i * HID, ln1b + i * HID, x);
    // FFN
    cvt(x, xh, U);
    tcvt(W1 + (long long)i * HID * FFDIM, wt, HID, FFDIM);
    gemmBig(xh, wt, hf, b1 + i * FFDIM, NTOK, FFDIM, HID, HID, HID, FFDIM, 1.f, 1, 1, 0, 0, 0, 0, 0, 0);
    gelu_kernel<<<(int)(((long long)NTOK * FFDIM + 255) / 256), 256, 0, stream>>>(hf, (long long)NTOK * FFDIM);
    cvt(hf, hh, (long long)NTOK * FFDIM);
    tcvt(W2 + (long long)i * FFDIM * HID, wt, FFDIM, HID);
    gemmBig(hh, wt, aof, b2 + i * HID, NTOK, HID, FFDIM, FFDIM, FFDIM, HID, 1.f, 1, 1, 0, 0, 0, 0, 0, 0);
    ln_kernel<<<NTOK, 256, 0, stream>>>(aof, x, ln2g + i * HID, ln2b + i * HID, x);
  }

  // ---- CRF heads
  head_gemm_kernel<<<(NTOK * 17 + 255) / 256, 256, 0, stream>>>(x, W_ele, b_ele, ele_em, 17);
  head_gemm_kernel<<<(NTOK * 7 + 255) / 256, 256, 0, stream>>>(x, W_role, b_role, role_em, 7);

  float* out = (float*)d_out;
  zero_loss_kernel<<<1, 1, 0, stream>>>(out);
  crf_llh_kernel<<<1, 512, 0, stream>>>(ele_em, ele_labels, smask, ele_start, ele_end, ele_trans, 17, out);
  crf_llh_kernel<<<1, 512, 0, stream>>>(role_em, role_labels, smask, role_start, role_end, role_trans, 7, out);
  crf_decode_kernel<<<1, 512, 0, stream>>>(ele_em, ele_start, ele_end, ele_trans, 17, hist, out + 1);
  crf_decode_kernel<<<1, 512, 0, stream>>>(role_em, role_start, role_end, role_trans, 7, hist, out + 1 + NTOK);
}
